// MemoryReader_29875792511366
// MI455X (gfx1250) — matrix operand order repaired and checked by reading, not compiled, not run on hardware
//
#include <hip/hip_runtime.h>

// ---------------------------------------------------------------------------
// MemoryReader (XMem-style) fused flash-softmax readout for gfx1250 (MI455X)
//   logits S = (2*mk^T qk - ||mk||^2) / sqrt(CK), softmax over memory dim,
//   out = mv @ P.  bf16 WMMA + Tensor Data Mover staging of the mv stream.
// ---------------------------------------------------------------------------

#define B_   4
#define CK_  64
#define T_   8
#define H_   30
#define W_   54
#define THW_ 12960            // T*H*W
#define NQ_  1620             // H*W
#define NQP_ 1664             // NQ padded to 26*64
#define CV_  512
#define MC_  32               // m-chunk (== one bf16 WMMA K)
#define NCHUNK_ (THW_ / MC_)  // 405
#define QT_  64               // queries per block

typedef __attribute__((ext_vector_type(16))) __bf16   v16bf;
typedef __attribute__((ext_vector_type(8)))  float    v8f;
typedef __attribute__((ext_vector_type(4)))  unsigned u32x4;
typedef __attribute__((ext_vector_type(8)))  unsigned u32x8;
typedef __attribute__((ext_vector_type(2)))  unsigned u32x2;
typedef __attribute__((ext_vector_type(4)))  float    f32x4;

union V16 { u32x4 u[2]; v16bf v; };

__device__ __forceinline__ unsigned short f2bf(float f) {
  unsigned x = __float_as_uint(f);
  unsigned r = (x + 0x7FFFu + ((x >> 16) & 1u)) >> 16;   // RNE
  return (unsigned short)r;
}
__device__ __forceinline__ unsigned pack2(float a, float b) {
  return (unsigned)f2bf(a) | ((unsigned)f2bf(b) << 16);
}
// Build a v16bf A/B operand from two 16-byte runs (callers use contiguous k = 16*half + s).
__device__ __forceinline__ v16bf ldv16(const unsigned short* p0,
                                       const unsigned short* p1) {
  V16 t;
  t.u[0] = *(const u32x4*)p0;
  t.u[1] = *(const u32x4*)p1;
  return t.v;
}
__device__ __forceinline__ unsigned rfl(unsigned v) {
  return (unsigned)__builtin_amdgcn_readfirstlane((int)v);
}

// ---------------------------------------------------------------------------
// TDM: issue a 2-D tensor_load_to_lds of bf16 data.
//   tile = tile_d1 rows x tile_d0 elems, row stride stride0 (elems),
//   compacted row-major into LDS at lds_off.  D# per ISA 08_async_tensor §8.
// ---------------------------------------------------------------------------
__device__ __forceinline__ void tdm_load_bf16_2d(unsigned lds_off,
                                                 unsigned long long gaddr,
                                                 unsigned tile_d0, unsigned tile_d1,
                                                 unsigned long long stride0) {
  const unsigned tensor_d0 = 0x40000000u, tensor_d1 = 0x40000000u;
  u32x4 g0;
  g0[0] = rfl(1u);                                        // count=1, user mode
  g0[1] = rfl(lds_off);                                   // lds_addr
  g0[2] = rfl((unsigned)gaddr);                           // global_addr lo
  g0[3] = rfl((unsigned)(gaddr >> 32) | (2u << 30));      // addr hi | type=2
  u32x8 g1;
  g1[0] = rfl(1u << 16);                                  // data_size=2B
  g1[1] = rfl((tensor_d0 & 0xFFFFu) << 16);               // tensor_dim0 lo16
  g1[2] = rfl((tensor_d0 >> 16) | ((tensor_d1 & 0xFFFFu) << 16));
  g1[3] = rfl((tensor_d1 >> 16) | (tile_d0 << 16));       // tile_dim0
  g1[4] = rfl(tile_d1);                                   // tile_dim1 (dim2=0)
  g1[5] = rfl((unsigned)stride0);                         // stride0 lo32
  g1[6] = rfl((unsigned)(stride0 >> 32) & 0xFFFFu);       // stride0 hi16
  g1[7] = rfl(0u);
  asm volatile("tensor_load_to_lds %0, %1" : : "s"(g0), "s"(g1) : "memory");
}

// ---------------------------------------------------------------------------
// Prep 1: mk [B,CK,THW] f32 -> mkT [B,THW,CK] bf16 scaled by 0.25,
//         abias[B,THW] = ||mk||^2 / 8   (logit = wmma - abias)
// ---------------------------------------------------------------------------
__global__ void mr_prep_mk(const float* __restrict__ mk,
                           unsigned short* __restrict__ mkT,
                           float* __restrict__ abias) {
  int t = blockIdx.x * 256 + threadIdx.x;
  if (t >= B_ * THW_) return;
  int b = t / THW_, m = t - b * THW_;
  const float* src = mk + (size_t)b * CK_ * THW_ + m;
  unsigned short* dst = mkT + (size_t)t * CK_;
  float sum = 0.f;
  for (int cb = 0; cb < CK_; cb += 8) {
    u32x4 pk;
    for (int i = 0; i < 8; i += 2) {
      float v0 = src[(size_t)(cb + i) * THW_];
      float v1 = src[(size_t)(cb + i + 1) * THW_];
      sum += v0 * v0 + v1 * v1;
      pk[i >> 1] = pack2(0.25f * v0, 0.25f * v1);
    }
    *(u32x4*)(dst + cb) = pk;
  }
  abias[t] = sum * 0.125f;
}

// ---------------------------------------------------------------------------
// Prep 2: qk [B,CK,NQ] f32 -> qkT [B,NQP,CK] bf16 (rows >= NQ zero padded)
// ---------------------------------------------------------------------------
__global__ void mr_prep_qk(const float* __restrict__ qk,
                           unsigned short* __restrict__ qkT) {
  int t = blockIdx.x * 256 + threadIdx.x;
  if (t >= B_ * NQP_) return;
  int b = t / NQP_, q = t - b * NQP_;
  unsigned short* dst = qkT + (size_t)t * CK_;
  if (q < NQ_) {
    const float* src = qk + (size_t)b * CK_ * NQ_ + q;
    for (int cb = 0; cb < CK_; cb += 8) {
      u32x4 pk;
      for (int i = 0; i < 8; i += 2)
        pk[i >> 1] = pack2(src[(size_t)(cb + i) * NQ_],
                           src[(size_t)(cb + i + 1) * NQ_]);
      *(u32x4*)(dst + cb) = pk;
    }
  } else {
    u32x4 z = {0u, 0u, 0u, 0u};
    for (int cb = 0; cb < CK_; cb += 8) *(u32x4*)(dst + cb) = z;
  }
}

// ---------------------------------------------------------------------------
// Prep 3: mv f32 -> bf16 (same [B,CV,THW] layout)
// ---------------------------------------------------------------------------
__global__ void mr_prep_mv(const float* __restrict__ mv,
                           unsigned short* __restrict__ mvh, int n4) {
  int i = blockIdx.x * 256 + threadIdx.x;
  if (i >= n4) return;
  f32x4 v = *(const f32x4*)(mv + (size_t)i * 4);
  u32x2 o;
  o[0] = pack2(v[0], v[1]);
  o[1] = pack2(v[2], v[3]);
  *(u32x2*)(mvh + (size_t)i * 4) = o;
}

// ---------------------------------------------------------------------------
// Fused kernel: one block = (batch b, 64-query tile). 8 waves.
//   Phase A (all waves, 1 tile each): S[32,64] logits via WMMA + online softmax
//   Phase B (all waves): acc[64v x 64q] = acc*f + WMMA(mv_lds, P)
//   mv chunks DMA'd by the Tensor Data Mover into wave-private LDS buffers,
//   double buffered and synchronized with TENSORcnt only.
// ---------------------------------------------------------------------------
__global__ void __launch_bounds__(256, 1)
mr_fused(const unsigned short* __restrict__ mkT,
         const unsigned short* __restrict__ qkT,
         const float* __restrict__ abias,
         const unsigned short* __restrict__ mvh,
         float* __restrict__ out) {
  const int b   = blockIdx.y;
  const int qt0 = blockIdx.x * QT_;
  const int tid = threadIdx.x;
  const int wave = tid >> 5, lane = tid & 31;
  const int half = lane >> 4, ln = lane & 15;
  const int mr = wave >> 2;        // phase-A m-row tile (0..1)
  const int qc = wave & 3;         // phase-A q-col tile (0..3)
  const int v0w = wave * 64;       // phase-B v-channel base (64 per wave)

  __shared__ unsigned short mvbuf[8][2][64 * MC_];   // wave-private TDM dest
  __shared__ unsigned short Pbuf[QT_ * MC_];         // [q][m] bf16 (GEMM2 B)
  __shared__ float tmax[2][QT_], tsum[2][QT_];
  __shared__ float runmax[QT_], runsum[QT_], fscale[QT_];

  if (tid < QT_) { runmax[tid] = -3.0e38f; runsum[tid] = 0.f; }

  // qk B-operands (K=0..31 and K=32..63), held in registers for all chunks.
  const unsigned short* qrow = qkT + ((size_t)b * NQP_ + qt0 + 16 * qc + ln) * CK_;
  const v16bf bq0 = ldv16(qrow + half * 16,      qrow + half * 16 + 8);
  const v16bf bq1 = ldv16(qrow + 32 + half * 16, qrow + 32 + half * 16 + 8);

  v8f acc[16];
  for (int i = 0; i < 16; ++i)
    for (int j = 0; j < 8; ++j) acc[i][j] = 0.f;

  // TDM double-buffer setup: this wave streams mv rows [v0w, v0w+64).
  const unsigned long long mv_gbase =
      (unsigned long long)(size_t)(mvh + ((size_t)b * CV_ + v0w) * THW_);
  const unsigned mv_lds0 = (unsigned)(size_t)&mvbuf[wave][0][0];
  const unsigned mv_lds1 = (unsigned)(size_t)&mvbuf[wave][1][0];
  // prime chunk 0
  tdm_load_bf16_2d(mv_lds0, mv_gbase, MC_, 64, (unsigned long long)THW_);

  __syncthreads();

  for (int chunk = 0; chunk < NCHUNK_; ++chunk) {
    const int m0 = chunk * MC_;

    // prefetch next mv chunk via TDM into the other buffer
    const int cn = (chunk + 1 < NCHUNK_) ? chunk + 1 : chunk;
    tdm_load_bf16_2d((cn & 1) ? mv_lds1 : mv_lds0,
                     mv_gbase + (unsigned long long)cn * MC_ * 2, MC_, 64,
                     (unsigned long long)THW_);
    // TDMs complete in order per wave: <=1 outstanding => chunk's data ready
    asm volatile("s_wait_tensorcnt 1" ::: "memory");

    // ---- Phase A: one 16x16 logit tile per wave ------------------------
    const unsigned short* arow =
        mkT + ((size_t)b * THW_ + m0 + 16 * mr + ln) * CK_;
    __builtin_prefetch(arow + MC_ * CK_, 0, 1);          // next chunk mkT
    v16bf a0 = ldv16(arow + half * 16,      arow + half * 16 + 8);
    v16bf a1 = ldv16(arow + 32 + half * 16, arow + 32 + half * 16 + 8);
    v8f s;
    for (int j = 0; j < 8; ++j) s[j] = 0.f;
    s = __builtin_amdgcn_wmma_f32_16x16x32_bf16(false, a0, false, bq0,
                                                (short)0, s, false, false);
    s = __builtin_amdgcn_wmma_f32_16x16x32_bf16(false, a1, false, bq1,
                                                (short)0, s, false, false);
    // bias: logit = S - ||mk||^2/8 ; row m = m0 + 16*mr + 8*half + j
    const float* ab = abias + (size_t)b * THW_ + m0 + 16 * mr + 8 * half;
    f32x4 ab0 = *(const f32x4*)ab, ab1 = *(const f32x4*)(ab + 4);
    for (int j = 0; j < 4; ++j) { s[j] -= ab0[j]; s[j + 4] -= ab1[j]; }

    // tile column max (16 rows) -> LDS
    float cm = s[0];
    for (int j = 1; j < 8; ++j) cm = fmaxf(cm, s[j]);
    cm = fmaxf(cm, __shfl_xor(cm, 16));
    if (lane < 16) tmax[mr][16 * qc + lane] = cm;
    __syncthreads();

    // new running max, P = exp(S - nm), tile column sum, P -> LDS (bf16)
    float rm = runmax[16 * qc + ln];
    float nm = fmaxf(rm, fmaxf(tmax[0][16 * qc + ln], tmax[1][16 * qc + ln]));
    float p[8], ts = 0.f;
    for (int j = 0; j < 8; ++j) { p[j] = __expf(s[j] - nm); ts += p[j]; }
    ts += __shfl_xor(ts, 16);
    if (lane < 16) tsum[mr][16 * qc + lane] = ts;
    unsigned* prow = (unsigned*)&Pbuf[(16 * qc + ln) * MC_ + 16 * mr + 8 * half];
    for (int j = 0; j < 4; ++j) prow[j] = pack2(p[2 * j], p[2 * j + 1]);
    __syncthreads();

    // softmax running-state update (one thread per query column)
    if (tid < QT_) {
      int q = tid;
      float rm2 = runmax[q];
      float nm2 = fmaxf(rm2, fmaxf(tmax[0][q], tmax[1][q]));
      float f = __expf(rm2 - nm2);
      runsum[q] = runsum[q] * f + tsum[0][q] + tsum[1][q];
      runmax[q] = nm2;
      fscale[q] = f;
    }
    __syncthreads();

    // ---- Phase B: acc = acc*f + mv_lds @ P -----------------------------
    float fq[4];
    v16bf bp[4];
    for (int qs = 0; qs < 4; ++qs) {
      fq[qs] = fscale[16 * qs + ln];
      const unsigned short* pr = &Pbuf[(16 * qs + ln) * MC_ + half * 16];
      bp[qs] = ldv16(pr, pr + 8);
    }
    // rescale only when the running max actually moved (wave-uniform branch)
    bool moved = (fq[0] != 1.f) | (fq[1] != 1.f) | (fq[2] != 1.f) | (fq[3] != 1.f);
    if (__any(moved)) {
      for (int vt = 0; vt < 4; ++vt)
        for (int qs = 0; qs < 4; ++qs)
          for (int j = 0; j < 8; ++j) acc[vt * 4 + qs][j] *= fq[qs];
    }
    const unsigned short* mvl = &mvbuf[wave][chunk & 1][0];
    for (int vt = 0; vt < 4; ++vt) {
      const unsigned short* avp = mvl + (16 * vt + ln) * MC_;
      v16bf av = ldv16(avp + half * 16, avp + half * 16 + 8);
      for (int qs = 0; qs < 4; ++qs) {
        const int idx = vt * 4 + qs;
        acc[idx] = __builtin_amdgcn_wmma_f32_16x16x32_bf16(
            false, av, false, bp[qs], (short)0, acc[idx], false, false);
      }
    }
  }

  // ---- finalize: out[b, v, q] = acc / runsum[q] ------------------------
  float inv[4];
  for (int qs = 0; qs < 4; ++qs) inv[qs] = 1.0f / runsum[16 * qs + ln];
  for (int vt = 0; vt < 4; ++vt)
    for (int qs = 0; qs < 4; ++qs) {
      int q = qt0 + 16 * qs + ln;
      if (q >= NQ_) continue;
      const int idx = vt * 4 + qs;
      for (int j = 0; j < 8; ++j) {
        int v = v0w + 16 * vt + 8 * half + j;
        out[((size_t)b * CV_ + v) * NQ_ + q] = acc[idx][j] * inv[qs];
      }
    }
}

// ---------------------------------------------------------------------------
extern "C" void kernel_launch(void* const* d_in, const int* in_sizes, int n_in,
                              void* d_out, int out_size, void* d_ws,
                              size_t ws_size, hipStream_t stream) {
  const float* mk = (const float*)d_in[0];
  const float* qk = (const float*)d_in[1];
  const float* mv = (const float*)d_in[2];
  float* out = (float*)d_out;

  // workspace carve-up (all 256B aligned)
  char* ws = (char*)d_ws;
  const size_t mvh_bytes  = (size_t)B_ * CV_ * THW_ * 2;   // 53,084,160
  const size_t mkT_bytes  = (size_t)B_ * THW_ * CK_ * 2;   //  6,635,520
  const size_t qkT_bytes  = (size_t)B_ * NQP_ * CK_ * 2;   //    851,968
  unsigned short* mvh   = (unsigned short*)ws;
  unsigned short* mkT   = (unsigned short*)(ws + mvh_bytes);
  unsigned short* qkT   = (unsigned short*)(ws + mvh_bytes + mkT_bytes);
  float*          abias = (float*)(ws + mvh_bytes + mkT_bytes + qkT_bytes);

  {
    int n = B_ * THW_;
    mr_prep_mk<<<(n + 255) / 256, 256, 0, stream>>>(mk, mkT, abias);
  }
  {
    int n = B_ * NQP_;
    mr_prep_qk<<<(n + 255) / 256, 256, 0, stream>>>(qk, qkT);
  }
  {
    int n4 = (B_ * CV_ * THW_) / 4;
    mr_prep_mv<<<(n4 + 255) / 256, 256, 0, stream>>>(mv, mvh, n4);
  }
  {
    dim3 grid((NQ_ + QT_ - 1) / QT_, B_);   // 26 x 4
    mr_fused<<<grid, 256, 0, stream>>>(mkT, qkT, abias, mvh, out);
  }
}
